// RelNet_5446018531683
// MI455X (gfx1250) — compile-verified, matrix operand order checked by reading, not run on hardware
//
#include <hip/hip_runtime.h>

// ---------------------------------------------------------------------------
// Types for WMMA (gfx1250, wave32)
// ---------------------------------------------------------------------------
typedef __bf16 v16bf __attribute__((ext_vector_type(16)));
typedef float  v8f   __attribute__((ext_vector_type(8)));

// Type used by the async global->LDS builtins (per hipcc diagnostic):
// "__attribute__((__vector_size__(4 * sizeof(int)))) int *"
typedef int b128v __attribute__((vector_size(16)));

union BfFrag {
    struct { uint4 lo, hi; } u;   // 2 x 16B contiguous chunks per lane
    v16bf v;
};

__device__ __forceinline__ unsigned short f2bf(float x) {
    unsigned int u = __float_as_uint(x);
    unsigned int r = u + 0x7FFFu + ((u >> 16) & 1u);   // round-to-nearest-even
    return (unsigned short)(r >> 16);
}
__device__ __forceinline__ float bf2f(unsigned short h) {
    return __uint_as_float(((unsigned int)h) << 16);
}

// Async global->LDS path availability (gfx1250). Fall back to plain LDS store.
#if defined(__HIP_DEVICE_COMPILE__) && __has_builtin(__builtin_amdgcn_global_load_async_to_lds_b128)
#define USE_ASYNC_LDS 1
#else
#define USE_ASYNC_LDS 0
#endif

__device__ __forceinline__ void wait_asynccnt0() {
#if defined(__HIP_DEVICE_COMPILE__)
#if __has_builtin(__builtin_amdgcn_s_wait_asynccnt)
    __builtin_amdgcn_s_wait_asynccnt(0);
#else
    asm volatile("s_wait_asynccnt 0x0" ::: "memory");
#endif
#endif
}

// ---------------------------------------------------------------------------
// Conv 3x3, stride 2, pad 1, + bias + relu  (NCHW, OIHW)
// ---------------------------------------------------------------------------
__global__ void conv_s2_relu(const float* __restrict__ x, const float* __restrict__ w,
                             const float* __restrict__ b, float* __restrict__ y,
                             int N, int Ci, int Hi, int Wi, int Co, int Ho, int Wo)
{
    size_t idx = (size_t)blockIdx.x * blockDim.x + threadIdx.x;
    size_t total = (size_t)N * Co * Ho * Wo;
    if (idx >= total) return;
    int ow = idx % Wo; size_t t = idx / Wo;
    int oh = t % Ho; t /= Ho;
    int co = t % Co; int n = t / Co;
    float acc = b[co];
    for (int ci = 0; ci < Ci; ++ci) {
        for (int kh = 0; kh < 3; ++kh) {
            int ih = oh * 2 - 1 + kh;
            if (ih < 0 || ih >= Hi) continue;
            for (int kw = 0; kw < 3; ++kw) {
                int iw = ow * 2 - 1 + kw;
                if (iw < 0 || iw >= Wi) continue;
                acc += x[(((size_t)n * Ci + ci) * Hi + ih) * Wi + iw] *
                       w[((co * Ci + ci) * 3 + kh) * 3 + kw];
            }
        }
    }
    y[idx] = fmaxf(acc, 0.0f);
}

// Per-channel batch statistics (deterministic block reduction); one block/channel.
__global__ void bn_stats(const float* __restrict__ y, const float* __restrict__ gamma,
                         float* __restrict__ mean_inv, int N, int C, int HW)
{
    int c = blockIdx.x;
    int tid = threadIdx.x;
    int cnt = N * HW;
    float s = 0.f, s2 = 0.f;
    for (int i = tid; i < cnt; i += blockDim.x) {
        int n = i / HW, p = i % HW;
        float v = y[((size_t)n * C + c) * HW + p];
        s += v; s2 += v * v;
    }
    __shared__ float sh[256], sh2[256];
    sh[tid] = s; sh2[tid] = s2;
    __syncthreads();
    for (int off = 128; off > 0; off >>= 1) {
        if (tid < off) { sh[tid] += sh[tid + off]; sh2[tid] += sh2[tid + off]; }
        __syncthreads();
    }
    if (tid == 0) {
        float m = sh[0] / (float)cnt;
        float var = sh2[0] / (float)cnt - m * m;
        mean_inv[c]     = m;
        mean_inv[C + c] = gamma[c] * rsqrtf(var + 1e-5f);
    }
}

__global__ void bn_apply(float* __restrict__ y, const float* __restrict__ mean_inv,
                         const float* __restrict__ beta, int C, int HW, size_t total)
{
    size_t idx = (size_t)blockIdx.x * blockDim.x + threadIdx.x;
    if (idx >= total) return;
    int c = (int)((idx / HW) % C);
    y[idx] = (y[idx] - mean_inv[c]) * mean_inv[C + c] + beta[c];
}

// ---------------------------------------------------------------------------
// Build pair feature matrix [512*25*25, 64] in bf16 (col 63 zero pad).
// row r = ((b*25 + kk)*25 + ii): [obj_ii(24+2) | obj_kk(24+2) | q(11) | 0]
// ---------------------------------------------------------------------------
__global__ void build_pairs(const float* __restrict__ x4, const float* __restrict__ qst,
                            unsigned short* __restrict__ P)
{
    size_t idx = (size_t)blockIdx.x * blockDim.x + threadIdx.x;
    const size_t total = (size_t)512 * 25 * 25 * 64;
    if (idx >= total) return;
    int col = idx & 63;
    size_t r = idx >> 6;
    int ii = r % 25; size_t t = r / 25;
    int kk = t % 25; int b = t / 25;
    float v = 0.0f;
    if (col < 24)       v = x4[((size_t)(b * 24 + col) * 25) + ii];
    else if (col == 24) v = (float)(ii / 5 - 2) * 0.5f;
    else if (col == 25) v = (float)(ii % 5 - 2) * 0.5f;
    else if (col < 50)  v = x4[((size_t)(b * 24 + (col - 26)) * 25) + kk];
    else if (col == 50) v = (float)(kk / 5 - 2) * 0.5f;
    else if (col == 51) v = (float)(kk % 5 - 2) * 0.5f;
    else if (col < 63)  v = qst[(size_t)(col - 52) * 512 + b];
    P[idx] = f2bf(v);
}

// Convert fp32 weight [N x K] to bf16 [N x Kpad] (zero padded).
__global__ void w_to_bf16(const float* __restrict__ W, unsigned short* __restrict__ Wbf,
                          int N, int K, int Kpad)
{
    int idx = blockIdx.x * blockDim.x + threadIdx.x;
    if (idx >= N * Kpad) return;
    int k = idx % Kpad, n = idx / Kpad;
    Wbf[idx] = (k < K) ? f2bf(W[(size_t)n * K + k]) : (unsigned short)0;
}

// ---------------------------------------------------------------------------
// WMMA GEMM: Out[M x 256](bf16) = relu( A[M x K](bf16) * W[256 x K]^T + bias )
// Block = 256 threads = 8 waves; block tile 32(M) x 256(N);
// each wave: 16(M) x 64(N) = 4 accumulators of v8f.
//
// A k-slices (32 rows x 32 k, 2KB) are staged into LDS via the CDNA5 async
// global->LDS path (ASYNCcnt), double-buffered; LDS row stride padded to
// 40 elements (80B) so per-lane 16B fragment reads are bank-conflict-free.
// Fragment layout per ISA 7.12.2 (16-bit A 16x32): lane holds row (lane&15),
// k-chunks [kb..kb+7] and [16+kb..16+kb+7], kb = (lane>>4)*8.
// B symmetric with N playing the role of M (W stored row-major [N][K]).
// ---------------------------------------------------------------------------
#define LDS_STRIDE 40   // elements per staged row (32 data + 8 pad)

template <int K>
__global__ __launch_bounds__(256) void gemm_bias_relu_bf16(
    const unsigned short* __restrict__ A,
    const unsigned short* __restrict__ W,
    const float* __restrict__ bias,
    unsigned short* __restrict__ Out,
    int M)
{
    __shared__ __align__(16) unsigned short tileA[2][32 * LDS_STRIDE];

    const int tid  = threadIdx.x;
    const int lane = tid & 31;
    const int wave = tid >> 5;             // 0..7
    const int mblk = blockIdx.x * 32;
    const int m0   = mblk + (wave & 1) * 16;
    const int n0   = (wave >> 1) * 64;
    const int row  = lane & 15;
    const int kb   = (lane >> 4) * 8;      // 0 or 8

    // Cooperative tile loader: tid 0..127 (waves 0..3) each move 16B.
    const int lr  = tid >> 2;              // staged row 0..31
    const int seg = tid & 3;               // 16B segment within the 32-k slice

    auto load_tile = [&](int kt, int buf) {
        if (tid < 128) {
            const unsigned short* g = A + (size_t)(mblk + lr) * K + kt + seg * 8;
            unsigned short* l = &tileA[buf][lr * LDS_STRIDE + seg * 8];
#if USE_ASYNC_LDS
            __builtin_amdgcn_global_load_async_to_lds_b128(
                (__attribute__((address_space(1))) b128v*)g,
                (__attribute__((address_space(3))) b128v*)l, 0, 0);
#else
            *(uint4*)l = *(const uint4*)g;
#endif
        }
    };

    v8f acc[4];
#pragma unroll
    for (int t = 0; t < 4; ++t) acc[t] = (v8f)(0.0f);

    load_tile(0, 0);

#pragma unroll
    for (int kt = 0; kt < K; kt += 32) {
        const int buf = (kt >> 5) & 1;
#if USE_ASYNC_LDS
        wait_asynccnt0();                  // loader waves drain ASYNCcnt
#endif
        __syncthreads();                   // staged tile visible to all waves

        BfFrag a;
        const unsigned short* ar = &tileA[buf][((wave & 1) * 16 + row) * LDS_STRIDE];
        a.u.lo = *(const uint4*)(ar + kb);
        a.u.hi = *(const uint4*)(ar + 16 + kb);

        if (kt + 32 < K) load_tile(kt + 32, buf ^ 1);   // prefetch next slice

#pragma unroll
        for (int t = 0; t < 4; ++t) {
            const unsigned short* wr = W + (size_t)(n0 + 16 * t + row) * K + kt;
            BfFrag bfr;
            bfr.u.lo = *(const uint4*)(wr + kb);
            bfr.u.hi = *(const uint4*)(wr + kb + 16);
            acc[t] = __builtin_amdgcn_wmma_f32_16x16x32_bf16(
                false, a.v, false, bfr.v, (short)0, acc[t], false, false);
        }
    }

    // C/D layout: lanes 0-15 -> M = r, lanes 16-31 -> M = 8 + r; N = lane&15.
    const int rowoff = (lane >> 4) * 8;
#pragma unroll
    for (int t = 0; t < 4; ++t) {
        int n = n0 + 16 * t + row;
        float bv = bias[n];
#pragma unroll
        for (int r = 0; r < 8; ++r) {
            float v = fmaxf(acc[t][r] + bv, 0.0f);
            Out[(size_t)(m0 + rowoff + r) * 256 + n] = f2bf(v);
        }
    }
}

// Sum 625 pair rows per batch element: xg[b][c] = sum_p H[b*625+p][c]
__global__ void sum_pairs(const unsigned short* __restrict__ H, float* __restrict__ xg)
{
    int b = blockIdx.x;
    int c = threadIdx.x;   // 256
    const unsigned short* p = H + (size_t)b * 625 * 256 + c;
    float s = 0.f;
    for (int i = 0; i < 625; ++i) s += bf2f(p[(size_t)i * 256]);
    xg[b * 256 + c] = s;
}

// Small dense layer: Y[M x N] = (relu?) (X[M x K] * W[N x K]^T + B)
__global__ void lin_kernel(const float* __restrict__ X, const float* __restrict__ W,
                           const float* __restrict__ B, float* __restrict__ Y,
                           int M, int N, int K, int do_relu)
{
    int idx = blockIdx.x * blockDim.x + threadIdx.x;
    if (idx >= M * N) return;
    int n = idx % N, m = idx / N;
    float a = B[n];
    const float* xr = X + (size_t)m * K;
    const float* wr = W + (size_t)n * K;
    for (int k = 0; k < K; ++k) a += xr[k] * wr[k];
    Y[idx] = do_relu ? fmaxf(a, 0.0f) : a;
}

__global__ void log_softmax10(const float* __restrict__ L, float* __restrict__ out)
{
    int b = blockIdx.x * blockDim.x + threadIdx.x;
    if (b >= 512) return;
    const float* l = L + b * 10;
    float mx = l[0];
    for (int i = 1; i < 10; ++i) mx = fmaxf(mx, l[i]);
    float s = 0.f;
    for (int i = 0; i < 10; ++i) s += expf(l[i] - mx);
    float ls = logf(s);
    for (int i = 0; i < 10; ++i) out[b * 10 + i] = l[i] - mx - ls;
}

// ---------------------------------------------------------------------------
// Launcher
// ---------------------------------------------------------------------------
extern "C" void kernel_launch(void* const* d_in, const int* in_sizes, int n_in,
                              void* d_out, int out_size, void* d_ws, size_t ws_size,
                              hipStream_t stream)
{
    (void)in_sizes; (void)n_in; (void)out_size; (void)ws_size;
    char* base = (char*)d_ws;

    // --- workspace layout (bytes, all 256B aligned) ---
    // conv chain (phase A):
    float* c1 = (float*)(base + 0);              // 512*24*40*40 f32 = 78,643,200 B
    float* c2 = (float*)(base + 78643200);       // 512*24*20*20 f32 = 19,660,800 B
    float* c3 = (float*)(base + 98304000);       // 512*24*10*10 f32 =  4,915,200 B
    float* c4 = (float*)(base + 103219200);      // 512*24*5*5  f32 =  1,228,800 B
    // overlays (phase B/C): pairs overwrite c1 region (dead); hA overwrites rest.
    unsigned short* pairs = (unsigned short*)(base + 0);          // 320000*64*2 = 40,960,000 B
    unsigned short* hA    = (unsigned short*)(base + 40960000);   // 320000*256*2 = 163,840,000 B
    unsigned short* hB    = (unsigned short*)(base + 204800000);  // 163,840,000 B
    // tail region (never overlapped):
    size_t tail = 368640000;
    float* stats[4];
    for (int l = 0; l < 4; ++l) stats[l] = (float*)(base + tail + (size_t)l * 256);
    unsigned short* wbf1 = (unsigned short*)(base + tail + 4096);          //  32,768 B
    unsigned short* wbf2 = (unsigned short*)(base + tail + 4096 + 32768);  // 131,072 B
    unsigned short* wbf3 = (unsigned short*)(base + tail + 4096 + 163840);
    unsigned short* wbf4 = (unsigned short*)(base + tail + 4096 + 294912);
    size_t tail2 = tail + 4096 + 425984;
    float* xg     = (float*)(base + tail2);            // 512*256*4 = 524,288 B
    float* f1o    = (float*)(base + tail2 + 524288);
    float* f2o    = (float*)(base + tail2 + 1048576);
    float* logits = (float*)(base + tail2 + 1572864);  // 512*10*4

    const float* img = (const float*)d_in[0];
    const float* qst = (const float*)d_in[1];

    // --- conv + bn chain ---
    struct { const float *x; float *y; int Ci, Hi, Wi, Ho, Wo; int wi; } L[4] = {
        { img, c1,  3, 80, 80, 40, 40,  2 },
        { c1,  c2, 24, 40, 40, 20, 20,  6 },
        { c2,  c3, 24, 20, 20, 10, 10, 10 },
        { c3,  c4, 24, 10, 10,  5,  5, 14 },
    };
    for (int l = 0; l < 4; ++l) {
        const float* cw = (const float*)d_in[L[l].wi + 0];
        const float* cb = (const float*)d_in[L[l].wi + 1];
        const float* bg = (const float*)d_in[L[l].wi + 2];
        const float* bb = (const float*)d_in[L[l].wi + 3];
        int HW = L[l].Ho * L[l].Wo;
        size_t total = (size_t)512 * 24 * HW;
        conv_s2_relu<<<dim3((unsigned)((total + 255) / 256)), 256, 0, stream>>>(
            L[l].x, cw, cb, L[l].y, 512, L[l].Ci, L[l].Hi, L[l].Wi, 24, L[l].Ho, L[l].Wo);
        bn_stats<<<24, 256, 0, stream>>>(L[l].y, bg, stats[l], 512, 24, HW);
        bn_apply<<<dim3((unsigned)((total + 255) / 256)), 256, 0, stream>>>(
            L[l].y, stats[l], bb, 24, HW, total);
    }

    // --- pair features (bf16, K padded to 64) ---
    {
        size_t total = (size_t)512 * 25 * 25 * 64;
        build_pairs<<<dim3((unsigned)((total + 255) / 256)), 256, 0, stream>>>(c4, qst, pairs);
    }

    // --- weight conversion to bf16 ---
    w_to_bf16<<<(256 * 64 + 255) / 256, 256, 0, stream>>>((const float*)d_in[18], wbf1, 256, 63, 64);
    w_to_bf16<<<(256 * 256 + 255) / 256, 256, 0, stream>>>((const float*)d_in[20], wbf2, 256, 256, 256);
    w_to_bf16<<<(256 * 256 + 255) / 256, 256, 0, stream>>>((const float*)d_in[22], wbf3, 256, 256, 256);
    w_to_bf16<<<(256 * 256 + 255) / 256, 256, 0, stream>>>((const float*)d_in[24], wbf4, 256, 256, 256);

    // --- g-MLP via WMMA: M = 320000 rows, N = 256 ---
    const int M = 512 * 25 * 25;
    const int nblk = M / 32;   // 10000
    gemm_bias_relu_bf16<64><<<nblk, 256, 0, stream>>>(pairs, wbf1, (const float*)d_in[19], hA, M);
    gemm_bias_relu_bf16<256><<<nblk, 256, 0, stream>>>(hA, wbf2, (const float*)d_in[21], hB, M);
    gemm_bias_relu_bf16<256><<<nblk, 256, 0, stream>>>(hB, wbf3, (const float*)d_in[23], hA, M);
    gemm_bias_relu_bf16<256><<<nblk, 256, 0, stream>>>(hA, wbf4, (const float*)d_in[25], hB, M);

    // --- reduce over 625 pairs ---
    sum_pairs<<<512, 256, 0, stream>>>(hB, xg);

    // --- f-MLP + log_softmax ---
    lin_kernel<<<(512 * 256 + 255) / 256, 256, 0, stream>>>(
        xg, (const float*)d_in[26], (const float*)d_in[27], f1o, 512, 256, 256, 1);
    lin_kernel<<<(512 * 256 + 255) / 256, 256, 0, stream>>>(
        f1o, (const float*)d_in[28], (const float*)d_in[29], f2o, 512, 256, 256, 1);
    lin_kernel<<<(512 * 10 + 255) / 256, 256, 0, stream>>>(
        f2o, (const float*)d_in[30], (const float*)d_in[31], logits, 512, 10, 256, 0);
    log_softmax10<<<2, 256, 0, stream>>>(logits, (float*)d_out);
}
